// TabEncoder_19310172962996
// MI455X (gfx1250) — matrix operand order repaired and checked by reading, not compiled, not run on hardware
//
#include <hip/hip_runtime.h>
#include <hip/hip_bf16.h>

// ---------------------------------------------------------------------------
// MI455X fused TabEncoder.
// GEMM1 [B,540]x[540,64] dominates compute (9.1 GFLOP); it runs on
// v_wmma_f32_16x16x32_f16 with f32 accumulation. Overall the kernel is
// gather-latency bound (~500MB @ 23.3TB/s => ~21us floor), so LDS is kept at
// 155,648B (< 160KB) to fit TWO workgroups per 320KB WGP: 16 waves/WGP for
// latency hiding during the random embedding gathers.
// ---------------------------------------------------------------------------

typedef _Float16 v8h  __attribute__((ext_vector_type(8)));
typedef _Float16 v16h __attribute__((ext_vector_type(16)));
typedef float    v8f  __attribute__((ext_vector_type(8)));

#define RPB        128          // rows per block (8 waves x 16-row WMMA tiles)
#define NT         26           // number of embedding tables
#define FUSED_PAD  544          // 540 padded to K-chunk multiple (17 x 32)
#define KCHUNKS1   17
#define W1_HALVES  (KCHUNKS1*4*32*16)   // 34816 packed halves
#define W2_HALVES  (2*2*32*16)          // 2048 packed halves

__device__ __constant__ int kDims[NT] = {
  32,32,32,32,32,32,32,32,32,32,32,32,21,21,14,14,9,9,6,6,4,4,4,4,4,4};
__device__ __constant__ int kCatOff[NT] = {   // prefix sums of kDims
  0,32,64,96,128,160,192,224,256,288,320,352,
  384,405,426,440,454,463,472,478,484,488,492,496,500,504};
__device__ __constant__ int kOffs[NT] = {     // row offsets into emb_flat
  0,1000001,1500002,1600003,1650004,1660005,1670006,1675007,1680008,
  1681009,1682010,1682511,1683012,1683113,1683214,1683265,1683316,
  1683337,1683358,1683369,1683380,1683386,1683392,1683396,1683400,1683403};

// ---------------------------------------------------------------------------
// Kernel 0: repack W1 (540x64) and W2 (64x32) f32 -> f16 WMMA B-fragments.
// Fragment (chunk c, ntile t): lane L holds 16 contiguous halves;
//   half i <-> element K = c*32 + i + (L&16), N = t*16 + (L&15).
// Main kernel then loads each fragment with two global_load_b128 per lane.
// ---------------------------------------------------------------------------
__global__ __launch_bounds__(256) void pack_weights_kernel(
    const float* __restrict__ W1, const float* __restrict__ W2,
    _Float16* __restrict__ wpack)
{
  int id = blockIdx.x * 256 + threadIdx.x;
  if (id < W1_HALVES) {
    int i    = id & 15;
    int lane = (id >> 4) & 31;
    int frag = id >> 9;
    int c = frag >> 2, t = frag & 3;
    int k = c * 32 + i + (lane & 16);
    int n = t * 16 + (lane & 15);
    float v = (k < 540) ? W1[k * 64 + n] : 0.0f;
    wpack[id] = (_Float16)v;
  } else if (id < W1_HALVES + W2_HALVES) {
    int id2  = id - W1_HALVES;
    int i    = id2 & 15;
    int lane = (id2 >> 4) & 31;
    int frag = id2 >> 9;
    int c = frag >> 1, t = frag & 1;
    int k = c * 32 + i + (lane & 16);
    int n = t * 16 + (lane & 15);
    wpack[id] = (_Float16)W2[k * 32 + n];
  }
}

__device__ __forceinline__ float silu(float x) {
  return x / (1.0f + __expf(-x));
}

__device__ __forceinline__ v16h cat16(v8h lo, v8h hi) {
  return __builtin_shufflevector(lo, hi, 0,1,2,3,4,5,6,7,8,9,10,11,12,13,14,15);
}

// ---------------------------------------------------------------------------
// Kernel 1: fused gather + cont-MLP + two WMMA GEMMs.
// ---------------------------------------------------------------------------
__global__ __launch_bounds__(256) void tab_encoder_kernel(
    const float* __restrict__ xc,    const int* __restrict__ xcat,
    const float* __restrict__ emb,   const float* __restrict__ Wc,
    const float* __restrict__ bc,    const float* __restrict__ lng,
    const float* __restrict__ lnb,   const float* __restrict__ b1,
    const float* __restrict__ b2,    const _Float16* __restrict__ wpack,
    float* __restrict__ out,         int nrows)
{
  extern __shared__ char smem[];
  _Float16* fused = (_Float16*)smem;                 // [RPB][544] f16 (139,264B)
  _Float16* zst   = fused + RPB * FUSED_PAD;         // [RPB][64]  f16 ( 16,384B)

  const int tid     = threadIdx.x;
  const int rowbase = blockIdx.x * RPB;

  // Zero the K-padding columns 540..543.
  for (int i = tid; i < RPB * 4; i += 256)
    fused[(i >> 2) * FUSED_PAD + 540 + (i & 3)] = (_Float16)0.0f;

  // ---------------- Phase 1a: cont MLP (Linear 64->32 + LN + SiLU) ---------
  // Wc (8KB) is shared by every block -> stays hot in WGP$/L2; reading it
  // directly (instead of staging in LDS) keeps the LDS footprint < 160KB so
  // two workgroups co-reside per WGP (16 waves for gather latency hiding).
  {
    const int row  = tid >> 1;          // 2 threads per row
    const int half = tid & 1;           // each owns 16 of 32 outputs
    const int grow = rowbase + row;
    if (grow < nrows) {
      float acc[16];
      #pragma unroll
      for (int n = 0; n < 16; n++) acc[n] = bc[half * 16 + n];
      const float4* x4 = (const float4*)(xc + (long)grow * 64);
      #pragma unroll 4
      for (int k4 = 0; k4 < 16; k4++) {
        float4 xv = x4[k4];
        float xk[4] = {xv.x, xv.y, xv.z, xv.w};
        #pragma unroll
        for (int j = 0; j < 4; j++) {
          const float* w = Wc + (k4 * 4 + j) * 32 + half * 16;
          #pragma unroll
          for (int n = 0; n < 16; n++) acc[n] = fmaf(xk[j], w[n], acc[n]);
        }
      }
      float s = 0.0f, sq = 0.0f;
      #pragma unroll
      for (int n = 0; n < 16; n++) { s += acc[n]; sq += acc[n] * acc[n]; }
      // partner lane (tid^1) holds the other 16 columns of this row (wave32)
      s  += __shfl_xor(s, 1, 32);
      sq += __shfl_xor(sq, 1, 32);
      float mu  = s * (1.0f / 32.0f);
      float var = sq * (1.0f / 32.0f) - mu * mu;
      float inv = rsqrtf(var + 1e-5f);
      _Float16* dst = fused + row * FUSED_PAD + half * 16;
      #pragma unroll
      for (int n = 0; n < 16; n++) {
        float v = (acc[n] - mu) * inv * lng[half * 16 + n] + lnb[half * 16 + n];
        dst[n] = (_Float16)silu(v);
      }
    }
  }

  // ---------------- Phase 1b: embedding gathers -> fused[:, 32:540] --------
  for (int job = tid; job < RPB * NT; job += 256) {
    int row = job / NT;
    int tbl = job - row * NT;
    int grow = rowbase + row;
    if (grow >= nrows) continue;
    int idx = xcat[(long)grow * NT + tbl];
    const float* src = emb + ((long)(kOffs[tbl] + idx)) * 32;
    float scale = (idx != 0) ? 1.0f : 0.0f;   // padding_idx = 0
    _Float16* dst = fused + row * FUSED_PAD + 32 + kCatOff[tbl];
    int dim = kDims[tbl];
    int d = 0;
    for (; d + 4 <= dim; d += 4) {
      float4 v = *(const float4*)(src + d);
      dst[d + 0] = (_Float16)(v.x * scale);
      dst[d + 1] = (_Float16)(v.y * scale);
      dst[d + 2] = (_Float16)(v.z * scale);
      dst[d + 3] = (_Float16)(v.w * scale);
    }
    for (; d < dim; d++) dst[d] = (_Float16)(src[d] * scale);
  }
  __syncthreads();

  // ---------------- Phase 2: per-wave 16-row WMMA tile ---------------------
  const int lane  = tid & 31;
  const int wave  = tid >> 5;
  const int g     = lane >> 4;      // half-wave group
  const int m     = lane & 15;      // row within tile (A) / column N (B,C,D)
  const int tile0 = wave * 16;

  // GEMM1: fused[16,544] x W1[544,64] -> z[16,64], bias + SiLU
  v8f acc[4];
  #pragma unroll
  for (int t = 0; t < 4; t++) {
    float bv = b1[t * 16 + m];
    #pragma unroll
    for (int r = 0; r < 8; r++) acc[t][r] = bv;
  }
  const _Float16* arow = fused + (tile0 + m) * FUSED_PAD;
  for (int c = 0; c < KCHUNKS1; c++) {
    v8h alo = *(const v8h*)(arow + c * 32 + 16 * g);       // K = 16g..16g+7
    v8h ahi = *(const v8h*)(arow + c * 32 + 16 * g + 8);   // K = 16g+8..
    v16h a = cat16(alo, ahi);
    const _Float16* bp = wpack + ((long)(c * 4) * 32 + lane) * 16;
    #pragma unroll
    for (int t = 0; t < 4; t++) {
      v8h blo = *(const v8h*)(bp + (long)t * 32 * 16);
      v8h bhi = *(const v8h*)(bp + (long)t * 32 * 16 + 8);
      v16h b = cat16(blo, bhi);
      acc[t] = __builtin_amdgcn_wmma_f32_16x16x32_f16(
          false, a, false, b, (short)0, acc[t], false, false);
    }
  }

  // SiLU + transpose z through per-wave LDS staging (LDS is in-order per wave).
  _Float16* zw = zst + tile0 * 64;          // [16][64]
  #pragma unroll
  for (int t = 0; t < 4; t++)
    #pragma unroll
    for (int r = 0; r < 8; r++)
      zw[(r + 8 * g) * 64 + t * 16 + m] = (_Float16)silu(acc[t][r]);

  // GEMM2: z[16,64] x W2[64,32] -> out[16,32], bias + SiLU
  v8f acc2[2];
  #pragma unroll
  for (int t = 0; t < 2; t++) {
    float bv = b2[t * 16 + m];
    #pragma unroll
    for (int r = 0; r < 8; r++) acc2[t][r] = bv;
  }
  const _Float16* a2row = zw + m * 64;
  const _Float16* w2p = wpack + W1_HALVES;
  #pragma unroll
  for (int c = 0; c < 2; c++) {
    v8h alo = *(const v8h*)(a2row + c * 32 + 16 * g);
    v8h ahi = *(const v8h*)(a2row + c * 32 + 16 * g + 8);
    v16h a = cat16(alo, ahi);
    #pragma unroll
    for (int t = 0; t < 2; t++) {
      const _Float16* bp = w2p + ((long)(c * 2 + t) * 32 + lane) * 16;
      v8h blo = *(const v8h*)(bp);
      v8h bhi = *(const v8h*)(bp + 8);
      v16h b = cat16(blo, bhi);
      acc2[t] = __builtin_amdgcn_wmma_f32_16x16x32_f16(
          false, a, false, b, (short)0, acc2[t], false, false);
    }
  }

  #pragma unroll
  for (int t = 0; t < 2; t++)
    #pragma unroll
    for (int r = 0; r < 8; r++) {
      int grow = rowbase + tile0 + r + 8 * g;
      if (grow < nrows)
        out[(long)grow * 32 + t * 16 + m] = silu(acc2[t][r]);
    }
}

// ---------------------------------------------------------------------------
extern "C" void kernel_launch(void* const* d_in, const int* in_sizes, int n_in,
                              void* d_out, int out_size, void* d_ws, size_t ws_size,
                              hipStream_t stream) {
  const float* xc   = (const float*)d_in[0];   // [B,64]
  const int*   xcat = (const int*)d_in[1];     // [B,26]
  const float* emb  = (const float*)d_in[2];   // [TOTAL_ROWS,32]
  const float* Wc   = (const float*)d_in[3];   // [64,32]
  const float* bc   = (const float*)d_in[4];   // [32]
  const float* lng  = (const float*)d_in[5];   // [32]
  const float* lnb  = (const float*)d_in[6];   // [32]
  const float* W1   = (const float*)d_in[7];   // [540,64]
  const float* b1   = (const float*)d_in[8];   // [64]
  const float* W2   = (const float*)d_in[9];   // [64,32]
  const float* b2   = (const float*)d_in[10];  // [32]
  float* out = (float*)d_out;

  int nrows = in_sizes[0] / 64;
  _Float16* wpack = (_Float16*)d_ws;           // 73,728 B used

  // Repack weights for WMMA B-fragment loads (re-run each call: deterministic).
  int pack_total = W1_HALVES + W2_HALVES;
  pack_weights_kernel<<<(pack_total + 255) / 256, 256, 0, stream>>>(W1, W2, wpack);

  // fused[128][544] f16 + zstage[128][64] f16 = 155,648B -> 2 workgroups/WGP.
  size_t lds_bytes = (size_t)RPB * FUSED_PAD * 2 + (size_t)RPB * 64 * 2;
  int nblocks = (nrows + RPB - 1) / RPB;
  tab_encoder_kernel<<<nblocks, 256, lds_bytes, stream>>>(
      xc, xcat, emb, Wc, bc, lng, lnb, b1, b2, wpack, out, nrows);
}
